// LinkPredictorGAT_79096117723245
// MI455X (gfx1250) — matrix operand order repaired and checked by reading, not compiled, not run on hardware
//
#include <hip/hip_runtime.h>
#include <hip/hip_bf16.h>

// ---------------------------------------------------------------------------
// Link predictor: out[e] = relu(concat(z[src[e]], z[dst[e]]) @ W1 + b1) @ W2 + b2
// z: [100000,256] f32, edges: [2,E] int, W1: [512,128], b1:[128], W2:[128,1], b2:[1]
//
// bf16 WMMA 16x16x32 (f32 accum). One wave = 16 edges (M=16), N=128 -> 8 acc tiles,
// K=512 -> 16 k-steps. W1 staged once per block into LDS (bf16, transposed [n][k],
// padded rows -> conflict-free ds_load_b128). An opaque per-tile zero (asm volatile)
// is folded into the LDS index so LICM cannot hoist the 128 W1 fragments out of the
// persistent loop (round 1: scratch spills) while keeping normal DS codegen
// (round 2's volatile-pointer attempt degraded to flat SCOPE_SYS loads).
// ---------------------------------------------------------------------------

typedef __attribute__((ext_vector_type(16))) __bf16        v16bf;
typedef __attribute__((ext_vector_type(8)))  float         v8f;
typedef __attribute__((ext_vector_type(4)))  float         v4f;
typedef __attribute__((ext_vector_type(4)))  unsigned int  v4u;

union BFrag {
    v4u   u[2];   // two 16-byte LDS loads
    v16bf bf;
};

#define KTOT      512      // 2*D
#define HDIM      128
#define LDS_KPAD  520      // 512 + 8 bf16 pad -> 1040B row stride (4-bank lane stride)
#define WAVES_PER_BLOCK 8

__global__ __launch_bounds__(256)
void linkpred_wmma_bf16_kernel(const float* __restrict__ z,
                               const int*   __restrict__ eidx,   // [2,E]
                               const float* __restrict__ W1,     // [512,128] row-major
                               const float* __restrict__ b1,     // [128]
                               const float* __restrict__ W2,     // [128]
                               const float* __restrict__ b2,     // [1]
                               float*       __restrict__ out,    // [E]
                               int E)
{
    extern __shared__ unsigned short smem[];   // W1^T as bf16: [128][LDS_KPAD]

    const int tid = threadIdx.x;

    // ---- Stage W1 -> LDS, bf16, transposed to [n][k] -------------------
    for (int i = tid; i < KTOT * HDIM; i += 256) {
        const int k = i >> 7;         // row of W1 (K index)
        const int n = i & (HDIM - 1); // col of W1 (N index)
        __bf16 v = (__bf16)W1[i];
        smem[n * LDS_KPAD + k] = __builtin_bit_cast(unsigned short, v);
    }
    __syncthreads();

    const int lane = tid & 31;
    const int wave = tid >> 5;
    const int m    = lane & 15;   // row-in-tile (A) == N-in-tile (B/C)
    const int half = lane >> 4;   // lane group: selects K slice / C row block

    const float b2v = b2[0];

    float b1v[8], w2v[8];
#pragma unroll
    for (int nt = 0; nt < 8; ++nt) {
        b1v[nt] = b1[nt * 16 + m];
        w2v[nt] = W2[nt * 16 + m];
    }

    const int numTiles = (E + 15) >> 4;

    for (int tile = blockIdx.x * WAVES_PER_BLOCK + wave; tile < numTiles;
         tile += gridDim.x * WAVES_PER_BLOCK) {

        // Opaque zero: compiler must treat it as loop-variant, so the LDS
        // B-fragment addresses below cannot be proven invariant -> no LICM,
        // ds_load_b128 stays inside the loop (no scratch spills).
        int zr;
        asm volatile("s_mov_b32 %0, 0" : "=s"(zr) : "s"(tile));

        const int eBase = tile << 4;
        int e = eBase + m;
        if (e >= E) e = E - 1;                  // clamp: keep EXEC all-ones for WMMA
        const long long srcRow = (long long)eidx[e];
        const long long dstRow = (long long)eidx[E + e];
        const float* __restrict__ zs = z + srcRow * 256;
        const float* __restrict__ zd = z + dstRow * 256;

        v8f acc[8];
        const v8f vz = {};
#pragma unroll
        for (int nt = 0; nt < 8; ++nt) acc[nt] = vz;

        // Bounded unroll: 4 k-steps in flight (keeps A-load pipelining within
        // the VGPR budget; full unroll caused scratch spills).
#pragma unroll 4
        for (int ks = 0; ks < 16; ++ks) {
            const int k0   = ks * 32;
            const int offA = k0 + half * 8;     // first 8-float chunk (K offA..offA+7)
            const int offB = offA + 16;         // second chunk (K offA+16..offA+23)

            // pair[j] = j<256 ? zs[j] : zd[j-256]; chunks never straddle 256.
            const float* p1 = (offA < 256) ? (zs + offA) : (zd + (offA - 256));
            const float* p2 = (offB < 256) ? (zs + offB) : (zd + (offB - 256));
            const v4f c0 = ((const v4f*)p1)[0];
            const v4f c1 = ((const v4f*)p1)[1];
            const v4f c2 = ((const v4f*)p2)[0];
            const v4f c3 = ((const v4f*)p2)[1];

            // A fragment: element i -> VGPR i/2, half i%2 (ISA 16-bit A layout)
            v16bf a;
            a[0]  = (__bf16)c0.x; a[1]  = (__bf16)c0.y;
            a[2]  = (__bf16)c0.z; a[3]  = (__bf16)c0.w;
            a[4]  = (__bf16)c1.x; a[5]  = (__bf16)c1.y;
            a[6]  = (__bf16)c1.z; a[7]  = (__bf16)c1.w;
            a[8]  = (__bf16)c2.x; a[9]  = (__bf16)c2.y;
            a[10] = (__bf16)c2.z; a[11] = (__bf16)c2.w;
            a[12] = (__bf16)c3.x; a[13] = (__bf16)c3.y;
            a[14] = (__bf16)c3.z; a[15] = (__bf16)c3.w;

            // B: lanes 0-15 take K k0..k0+7 and k0+16..k0+23, lanes 16-31 take +8.
            // zr == 0 always, but defeats loop-invariance proofs.
            const int kb = k0 + half * 8 + zr;

#pragma unroll
            for (int nt = 0; nt < 8; ++nt) {
                const v4u* bp = (const v4u*)(smem + (nt * 16 + m) * LDS_KPAD + kb);
                BFrag b;
                b.u[0] = bp[0];
                b.u[1] = bp[2];
                acc[nt] = __builtin_amdgcn_wmma_f32_16x16x32_bf16(
                    /*neg_a=*/false, a,
                    /*neg_b=*/false, b.bf,
                    /*c_mod=*/(short)0, acc[nt],
                    /*reuse_a=*/false, /*reuse_b=*/false);
            }
        }

        // ---- Epilogue: relu(acc + b1) * W2, reduce over N ---------------
        // C layout: lane holds column n = (lane&15); VGPR r holds row r (lanes 0-15)
        // or row r+8 (lanes 16-31).
        float part[8];
#pragma unroll
        for (int r = 0; r < 8; ++r) part[r] = 0.0f;

#pragma unroll
        for (int nt = 0; nt < 8; ++nt) {
#pragma unroll
            for (int r = 0; r < 8; ++r) {
                float h = acc[nt][r] + b1v[nt];
                h = (h > 0.0f) ? h : 0.0f;
                part[r] += h * w2v[nt];
            }
        }

        // Reduce over the 16 N-lanes of each half-wave.
#pragma unroll
        for (int r = 0; r < 8; ++r) {
            part[r] += __shfl_xor(part[r], 1, 32);
            part[r] += __shfl_xor(part[r], 2, 32);
            part[r] += __shfl_xor(part[r], 4, 32);
            part[r] += __shfl_xor(part[r], 8, 32);
        }

        if (m == 0) {
#pragma unroll
            for (int r = 0; r < 8; ++r) {
                const int eo = eBase + half * 8 + r;
                if (eo < E) out[eo] = part[r] + b2v;
            }
        }
    }
}

extern "C" void kernel_launch(void* const* d_in, const int* in_sizes, int n_in,
                              void* d_out, int out_size, void* d_ws, size_t ws_size,
                              hipStream_t stream) {
    const float* z   = (const float*)d_in[0];
    const int*   ei  = (const int*)  d_in[1];
    const float* W1  = (const float*)d_in[2];
    const float* b1  = (const float*)d_in[3];
    const float* W2  = (const float*)d_in[4];
    const float* b2  = (const float*)d_in[5];
    float*       out = (float*)d_out;

    const int E = in_sizes[1] / 2;
    const int numTiles = (E + 15) / 16;
    int blocks = (numTiles + WAVES_PER_BLOCK - 1) / WAVES_PER_BLOCK;
    if (blocks > 1024) blocks = 1024;   // persistent grid; amortize W1 LDS staging
    if (blocks < 1)    blocks = 1;

    const size_t shmem = (size_t)HDIM * LDS_KPAD * sizeof(unsigned short); // 130 KB

    linkpred_wmma_bf16_kernel<<<dim3(blocks), dim3(256), shmem, stream>>>(
        z, ei, W1, b1, W2, b2, out, E);
}
